// Linear4bit_5214090297851
// MI455X (gfx1250) — compile-verified, matrix operand order checked by reading, not run on hardware
//
#include <hip/hip_runtime.h>
#include <hip/hip_bf16.h>

typedef __attribute__((ext_vector_type(16))) _Float16 v16h;
typedef __attribute__((ext_vector_type(8)))  _Float16 v8h;
typedef __attribute__((ext_vector_type(2)))  _Float16 v2h;
typedef __attribute__((ext_vector_type(8)))  float    v8f;
typedef __attribute__((ext_vector_type(4)))  int      v4i;

#define HAD_SCALE 0.08838834764831843f   // 2^-3.5

// ---- async global->LDS copy (CDNA5), with portable fallback -----------------
#if defined(__has_builtin)
#if __has_builtin(__builtin_amdgcn_global_load_async_to_lds_b128)
#define HAVE_ASYNC_LDS 1
#endif
#endif

typedef __attribute__((address_space(1))) v4i* gv4i_p;   // global int4*
typedef __attribute__((address_space(3))) v4i* lv4i_p;   // LDS int4*

__device__ __forceinline__ void async_cp16(const void* g, void* l)
{
#if defined(HAVE_ASYNC_LDS)
    __builtin_amdgcn_global_load_async_to_lds_b128(
        (gv4i_p)(g), (lv4i_p)(l), 0, 0);
#else
    *(uint4*)l = *(const uint4*)g;   // sync fallback: load + ds_store
#endif
}

__device__ __forceinline__ void wait_async0()
{
#if defined(HAVE_ASYNC_LDS)
#if __has_builtin(__builtin_amdgcn_s_wait_asynccnt)
    __builtin_amdgcn_s_wait_asynccnt(0);
#else
    asm volatile("s_wait_asynccnt 0x0" ::: "memory");
#endif
#endif
}

// ---------------------------------------------------------------------------
// Kernel 1: unpack int4 nibbles (one pair per int32) -> f16 weight panel
// ---------------------------------------------------------------------------
__global__ __launch_bounds__(256) void unpack_w4_kernel(
    const int* __restrict__ packed, _Float16* __restrict__ wf, int total)
{
    int i = blockIdx.x * 256 + threadIdx.x;
    if (i >= total) return;
    int p = packed[i];
    v2h v;
    v[0] = (_Float16)(float)(p & 0xF);
    v[1] = (_Float16)(float)((p >> 4) & 0xF);
    *(v2h*)&wf[(size_t)2 * i] = v;
}

// ---------------------------------------------------------------------------
// Kernel 2: blockwise FWHT-128 (wave32: 4 elems/lane; strides 32/64 in regs,
// strides 1..16 via shfl_xor), scale by 2^-3.5, cast f16, per-row sum.
// ---------------------------------------------------------------------------
__global__ __launch_bounds__(256) void had_rowsum_kernel(
    const float* __restrict__ x, _Float16* __restrict__ a,
    float* __restrict__ rowsum, int K)
{
    const int m    = blockIdx.x;
    const int tid  = threadIdx.x;
    const int lane = tid & 31;
    const int wave = tid >> 5;
    const int nblk = K >> 7;

    float lsum = 0.0f;

    for (int hb = wave; hb < nblk; hb += 8) {
        const size_t base = (size_t)m * K + (size_t)hb * 128;
        float v[4];
#pragma unroll
        for (int k = 0; k < 4; ++k) v[k] = x[base + lane + 32 * k];

        { float t0 = v[0] + v[1], t1 = v[0] - v[1];
          float t2 = v[2] + v[3], t3 = v[2] - v[3];
          v[0] = t0; v[1] = t1; v[2] = t2; v[3] = t3; }
        { float t0 = v[0] + v[2], t2 = v[0] - v[2];
          float t1 = v[1] + v[3], t3 = v[1] - v[3];
          v[0] = t0; v[1] = t1; v[2] = t2; v[3] = t3; }
#pragma unroll
        for (int st = 16; st >= 1; st >>= 1) {
#pragma unroll
            for (int k = 0; k < 4; ++k) {
                float p = __shfl_xor(v[k], st, 32);
                v[k] = (lane & st) ? (p - v[k]) : (v[k] + p);
            }
        }
#pragma unroll
        for (int k = 0; k < 4; ++k) {
            float y = v[k] * HAD_SCALE;
            lsum += y;
            a[base + lane + 32 * k] = (_Float16)y;
        }
    }

#pragma unroll
    for (int off = 16; off >= 1; off >>= 1) lsum += __shfl_xor(lsum, off, 32);
    __shared__ float ws[8];
    if (lane == 0) ws[wave] = lsum;
    __syncthreads();
    if (tid == 0) {
        float t = 0.0f;
#pragma unroll
        for (int i = 0; i < 8; ++i) t += ws[i];
        rowsum[m] = t;
    }
}

// ---------------------------------------------------------------------------
// Kernel 3: f16 WMMA GEMM, block tile 128(M) x 256(N), 8 waves (2x4),
// wave tile 64x64 (4x4 WMMA frags), K-step 64, double-buffered LDS fed by
// async global->LDS copies (one barrier per K-step).
// out[m][n] = scale[n]*(A.W) + add[n]*rowsum[m] + bias[n]
// ---------------------------------------------------------------------------
__device__ __forceinline__ v16h lds_frag(const _Float16* p0, const _Float16* p1)
{
    v8h lo = *(const v8h*)p0;   // ds_load_b128
    v8h hi = *(const v8h*)p1;   // ds_load_b128
    v16h r;
#pragma unroll
    for (int i = 0; i < 8; ++i) { r[i] = lo[i]; r[i + 8] = hi[i]; }
    return r;
}

__global__ __launch_bounds__(256) void gemm_w4_kernel(
    const _Float16* __restrict__ A,   // [M][K]
    const _Float16* __restrict__ W,   // [N][K]
    const float* __restrict__ scale,
    const float* __restrict__ addv,
    const float* __restrict__ bias,
    const float* __restrict__ rowsum,
    float* __restrict__ out,          // [M][N]
    int M, int N, int K)
{
    __shared__ _Float16 As[2][128][72];   // 72 = 64 + 8 pad (rows stay 16B-aligned)
    __shared__ _Float16 Bs[2][256][72];

    const int tid   = threadIdx.x;
    const int lane  = tid & 31;
    const int wave  = tid >> 5;
    const int wm    = wave & 1;        // 2 strips of 64 rows
    const int wn    = wave >> 1;       // 4 strips of 64 cols
    const int lrow  = lane & 15;
    const int lhalf = lane >> 4;

    const int blockN = blockIdx.x * 256;
    const int blockM = blockIdx.y * 128;

    v8f acc[4][4] = {};

    // stage K-tile k0 into buffer buf (1024 A-chunks + 2048 B-chunks of 16B)
    auto stage = [&](int buf, int k0) {
#pragma unroll
        for (int i = 0; i < 4; ++i) {
            int c = tid + 256 * i, row = c >> 3, col = (c & 7) * 8;
            async_cp16(&A[(size_t)(blockM + row) * K + k0 + col],
                       &As[buf][row][col]);
        }
#pragma unroll
        for (int i = 0; i < 8; ++i) {
            int c = tid + 256 * i, row = c >> 3, col = (c & 7) * 8;
            async_cp16(&W[(size_t)(blockN + row) * K + k0 + col],
                       &Bs[buf][row][col]);
        }
    };

    stage(0, 0);
    wait_async0();
    __syncthreads();

    int buf = 0;
    for (int k0 = 0; k0 < K; k0 += 64, buf ^= 1) {
        if (k0 + 64 < K) stage(buf ^ 1, k0 + 64);

#pragma unroll
        for (int kk = 0; kk < 64; kk += 32) {
            v16h afr[4], bfr[4];
#pragma unroll
            for (int mt = 0; mt < 4; ++mt) {
                int r = wm * 64 + mt * 16 + lrow;
                afr[mt] = lds_frag(&As[buf][r][kk + lhalf * 8],
                                   &As[buf][r][kk + 16 + lhalf * 8]);
            }
#pragma unroll
            for (int nt = 0; nt < 4; ++nt) {
                int r = wn * 64 + nt * 16 + lrow;
                bfr[nt] = lds_frag(&Bs[buf][r][kk + lhalf * 8],
                                   &Bs[buf][r][kk + 16 + lhalf * 8]);
            }
#pragma unroll
            for (int mt = 0; mt < 4; ++mt)
#pragma unroll
                for (int nt = 0; nt < 4; ++nt)
                    acc[mt][nt] = __builtin_amdgcn_wmma_f32_16x16x32_f16(
                        false, afr[mt], false, bfr[nt],
                        (short)0, acc[mt][nt], false, false);
        }

        wait_async0();        // next-tile copies landed
        __syncthreads();      // everyone done reading buf / writing buf^1
    }

    // Epilogue: C layout -> lane holds N = lrow, VGPR j holds M = j + 8*lhalf
#pragma unroll
    for (int mt = 0; mt < 4; ++mt) {
        const int mbase = blockM + wm * 64 + mt * 16 + lhalf * 8;
        float rs[8];
#pragma unroll
        for (int j = 0; j < 8; ++j) rs[j] = rowsum[mbase + j];
#pragma unroll
        for (int nt = 0; nt < 4; ++nt) {
            const int n  = blockN + wn * 64 + nt * 16 + lrow;
            const float sc = scale[n];
            const float ad = addv[n];
            const float bs = bias[n];
#pragma unroll
            for (int j = 0; j < 8; ++j) {
                const size_t o = (size_t)(mbase + j) * N + n;
                out[o] = acc[mt][nt][j] * sc + rs[j] * ad + bs;
            }
        }
    }
}

// ---------------------------------------------------------------------------
extern "C" void kernel_launch(void* const* d_in, const int* in_sizes, int n_in,
                              void* d_out, int out_size, void* d_ws, size_t ws_size,
                              hipStream_t stream)
{
    const float* x    = (const float*)d_in[0];
    const int*   wpk  = (const int*)d_in[1];
    const float* wsc  = (const float*)d_in[2];
    const float* wad  = (const float*)d_in[3];
    const float* bias = (const float*)d_in[4];
    float* out = (float*)d_out;

    const int OUT = in_sizes[2];                          // 4096
    const int IN  = (int)((2LL * in_sizes[1]) / OUT);     // 4096
    const int M   = (int)(in_sizes[0] / IN);              // 8192

    char* ws = (char*)d_ws;
    _Float16* a_f16 = (_Float16*)ws;
    _Float16* w_f16 = (_Float16*)(ws + (size_t)M * IN * sizeof(_Float16));
    float* rowsum   = (float*)(ws + (size_t)M * IN * sizeof(_Float16)
                                  + (size_t)OUT * IN * sizeof(_Float16));

    const int totalPacked = OUT * (IN / 2);
    unpack_w4_kernel<<<(totalPacked + 255) / 256, 256, 0, stream>>>(
        wpk, w_f16, totalPacked);

    had_rowsum_kernel<<<M, 256, 0, stream>>>(x, a_f16, rowsum, IN);

    dim3 grid(OUT / 256, M / 128);
    gemm_w4_kernel<<<grid, 256, 0, stream>>>(
        a_f16, w_f16, wsc, wad, bias, rowsum, out, M, OUT, IN);
}
